// resgnn_5394478923809
// MI455X (gfx1250) — matrix operand order repaired and checked by reading, compiled, not run on hardware
//
#include <hip/hip_runtime.h>
#include <hip/hip_bf16.h>

#define DEV_INLINE __device__ __forceinline__

typedef __attribute__((ext_vector_type(16))) __bf16 v16bf;
typedef __attribute__((ext_vector_type(8)))  __bf16 v8bf;
typedef __attribute__((ext_vector_type(8)))  float  v8f;

constexpr int NN = 100000;   // nodes
constexpr int EE = 1600000;  // edges
constexpr int HH = 128;      // hidden = features
constexpr int CC = 10;       // classes
constexpr int KB = 4;        // weight banks
constexpr int DEPTH_ = 8;
constexpr int MT = 5;        // row tiles per block in gemm128 (6250 % 5 == 0)

// ---------------------------------------------------------------- helpers
DEV_INLINE unsigned short bfbits(float f) {
  unsigned u = __builtin_bit_cast(unsigned, f);
  return (unsigned short)((u + 0x7FFFu + ((u >> 16) & 1u)) >> 16);
}
DEV_INLINE __bf16 f2bf(float f) { return __builtin_bit_cast(__bf16, bfbits(f)); }

struct BfPair { v8bf lo, hi; };
DEV_INLINE v16bf combine(v8bf lo, v8bf hi) {
  BfPair p{lo, hi};
  return __builtin_bit_cast(v16bf, p);
}

struct AFrags { v16bf f[4]; };
// Load the 4 K-chunk A fragments of one 16-row tile for this lane.
// Lane holds row M=(lane&15); K set = ka+{0..7} U ka+{16..23}, ka = kk*32 + 8*(lane>=16).
DEV_INLINE AFrags loadA(const __bf16* __restrict__ arow, int koffLane) {
  AFrags r;
#pragma unroll
  for (int kk = 0; kk < 4; ++kk) {
    const int ka = kk * 32 + koffLane;
    r.f[kk] = combine(*(const v8bf*)(arow + ka), *(const v8bf*)(arow + ka + 16));
  }
  return r;
}

// ------------------------------------------------------- fp32 -> bf16 pass
__global__ void act2bf(const float* __restrict__ src, __bf16* __restrict__ dst, int n4) {
  int i = blockIdx.x * blockDim.x + threadIdx.x;
  if (i >= n4) return;
  float4 v = ((const float4*)src)[i];
  unsigned p0 = (unsigned)bfbits(v.x) | ((unsigned)bfbits(v.y) << 16);
  unsigned p1 = (unsigned)bfbits(v.z) | ((unsigned)bfbits(v.w) << 16);
  ((uint2*)dst)[i] = make_uint2(p0, p1);
}

// W2 [128,10] -> padded bf16 [128,16]
__global__ void conv_w2pad(const float* __restrict__ W2, __bf16* __restrict__ dst) {
  int i = blockIdx.x * blockDim.x + threadIdx.x;
  if (i >= HH * 16) return;
  int r = i >> 4, c = i & 15;
  dst[i] = (c < CC) ? f2bf(W2[r * CC + c]) : f2bf(0.0f);
}

// ------------------------------------------------------------ degree/norm
__global__ void deg_init(float* __restrict__ deg, int n) {
  int i = blockIdx.x * blockDim.x + threadIdx.x;
  if (i < n) deg[i] = 1.0f;  // self loop
}
__global__ void deg_edges(const int* __restrict__ dst, float* __restrict__ deg, int e) {
  int i = blockIdx.x * blockDim.x + threadIdx.x;
  if (i < e) atomicAdd(&deg[dst[i]], 1.0f);
}
__global__ void make_dinv(float* __restrict__ deg, int n) {
  int i = blockIdx.x * blockDim.x + threadIdx.x;
  if (i < n) deg[i] = __frsqrt_rn(deg[i]);   // deg >= 1 always
}

// ------------------------------------------------- bf16 WMMA GEMM, N=128
// out[M,128] = act( Abf[M,128] @ Bw[128,128] + bias )
// block = 8 waves (one column tile each), MT row tiles per block.
// B fragments resident in registers; A double-buffered across M tiles.
__global__ void gemm128(const __bf16* __restrict__ Abf, const __bf16* __restrict__ Bw,
                        const float* __restrict__ bias, float* __restrict__ out,
                        int relu) {
  const int lane = threadIdx.x & 31;
  const int wave = threadIdx.x >> 5;        // 8 waves -> 8 column tiles
  const int n0 = wave * 16;
  const int mBase0 = blockIdx.x * (16 * MT);
  const int mrow = lane & 15;
  const int koffLane = (lane >> 4) * 8;     // A-layout half-wave K shift

  // ---- preload B fragments: lane holds B col n0+(lane&15), K = kk*32+koffLane+{0..7,16..23}
  v16bf bfr[4];
#pragma unroll
  for (int kk = 0; kk < 4; ++kk) {
    const __bf16* __restrict__ bcol = Bw + (size_t)(kk * 32 + koffLane) * HH + n0 + mrow;
    v16bf t;
#pragma unroll
    for (int s = 0; s < 8; ++s) { t[s] = bcol[s * HH]; t[s + 8] = bcol[(s + 16) * HH]; }
    bfr[kk] = t;
  }

  const int n = n0 + mrow;
  const float bv = bias ? bias[n] : 0.0f;
  const __bf16* __restrict__ arow0 = Abf + (size_t)(mBase0 + mrow) * HH;

  // prime the pipeline: A fragments of tile 0
  AFrags cur = loadA(arow0, koffLane);

#pragma unroll
  for (int mt = 0; mt < MT; ++mt) {
    // issue next tile's loads before computing on current tile
    AFrags nxt;
    if (mt + 1 < MT)
      nxt = loadA(arow0 + (size_t)(mt + 1) * 16 * HH, koffLane);
    if (mt + 2 < MT)
      __builtin_prefetch(arow0 + (size_t)(mt + 2) * 16 * HH, 0, 3);

    v8f acc = {};
#pragma unroll
    for (int kk = 0; kk < 4; ++kk)
      acc = __builtin_amdgcn_wmma_f32_16x16x32_bf16(false, cur.f[kk], false, bfr[kk],
                                                    (short)0, acc, false, false);

    const int mB = mBase0 + mt * 16 + ((lane >> 4) << 3);
#pragma unroll
    for (int r = 0; r < 8; ++r) {
      float v = acc[r] + bv;
      if (relu) v = fmaxf(v, 0.0f);
      out[(size_t)(mB + r) * HH + n] = v;
    }
    if (mt + 1 < MT) cur = nxt;
  }
}

// ------------------------------------- final GEMM: [M,128] @ [128,16pad] -> [M,10]
__global__ void gemm_out(const __bf16* __restrict__ Abf, const __bf16* __restrict__ Bw,
                         const float* __restrict__ bias, float* __restrict__ out) {
  const int lane = threadIdx.x & 31;
  const int wave = threadIdx.x >> 5;
  const int rowBase = (blockIdx.x * 8 + wave) * 16;
  if (rowBase >= NN) return;                 // uniform per wave -> EXEC stays full
  const int mrow = lane & 15;
  const int koffLane = (lane >> 4) * 8;
  const __bf16* __restrict__ arow = Abf + (size_t)(rowBase + mrow) * HH;

  // load all fragments first, then run the WMMA chain
  AFrags a = loadA(arow, koffLane);
  v16bf b[4];
#pragma unroll
  for (int kk = 0; kk < 4; ++kk) {
    const __bf16* __restrict__ bcol = Bw + (size_t)(kk * 32 + koffLane) * 16 + mrow;
    v16bf t;
#pragma unroll
    for (int s = 0; s < 8; ++s) { t[s] = bcol[s * 16]; t[s + 8] = bcol[(s + 16) * 16]; }
    b[kk] = t;
  }
  v8f acc = {};
#pragma unroll
  for (int kk = 0; kk < 4; ++kk)
    acc = __builtin_amdgcn_wmma_f32_16x16x32_bf16(false, a.f[kk], false, b[kk],
                                                  (short)0, acc, false, false);

  const int nc = mrow;
  if (nc < CC) {
    const float bvv = bias[nc];
    const int mB = rowBase + ((lane >> 4) << 3);
#pragma unroll
    for (int r = 0; r < 8; ++r)
      out[(size_t)(mB + r) * CC + nc] = acc[r] + bvv;
  }
}

// ---------------------------------------------------------- aggregation
// agg = tmp * dinv^2  (self loop), one thread per float4
__global__ void agg_init(const float* __restrict__ tmp, const float* __restrict__ dinv,
                         float* __restrict__ agg) {
  int i = blockIdx.x * blockDim.x + threadIdx.x;
  if (i >= NN * HH / 4) return;
  int node = i >> 5;                       // 32 float4 per row
  float d = dinv[node];
  float s = d * d;
  float4 v = ((const float4*)tmp)[i];
  ((float4*)agg)[i] = make_float4(v.x * s, v.y * s, v.z * s, v.w * s);
}

// one wave per edge: agg[dst] += tmp[src] * (dinv[src]*dinv[dst])
__global__ void edge_agg(const float* __restrict__ tmp, const int* __restrict__ src,
                         const int* __restrict__ dst, const float* __restrict__ dinv,
                         float* __restrict__ agg, int e) {
  int w = (blockIdx.x * blockDim.x + threadIdx.x) >> 5;
  int lane = threadIdx.x & 31;
  if (w >= e) return;
  int s = src[w], t = dst[w];
  float nrm = dinv[s] * dinv[t];
  float4 v = ((const float4*)(tmp + (size_t)s * HH))[lane];
  float* ad = agg + (size_t)t * HH + lane * 4;
  atomicAdd(ad + 0, v.x * nrm);
  atomicAdd(ad + 1, v.y * nrm);
  atomicAdd(ad + 2, v.z * nrm);
  atomicAdd(ad + 3, v.w * nrm);
}

// h += relu(agg + bb_k), one thread per float4
__global__ void residual(float* __restrict__ h, const float* __restrict__ agg,
                         const float* __restrict__ bbk) {
  int i = blockIdx.x * blockDim.x + threadIdx.x;
  if (i >= NN * HH / 4) return;
  float4 a = ((const float4*)agg)[i];
  float4 b = ((const float4*)bbk)[i & 31];
  float4 hv = ((float4*)h)[i];
  hv.x += fmaxf(a.x + b.x, 0.0f);
  hv.y += fmaxf(a.y + b.y, 0.0f);
  hv.z += fmaxf(a.z + b.z, 0.0f);
  hv.w += fmaxf(a.w + b.w, 0.0f);
  ((float4*)h)[i] = hv;
}

// -------------------------------------------------------------- layernorm
__global__ void layernorm(const float* __restrict__ h, const float* __restrict__ gamma,
                          const float* __restrict__ beta, float* __restrict__ out,
                          int nrows) {
  int w = (blockIdx.x * blockDim.x + threadIdx.x) >> 5;
  int lane = threadIdx.x & 31;
  if (w >= nrows) return;
  float4 v = ((const float4*)(h + (size_t)w * HH))[lane];
  float s = v.x + v.y + v.z + v.w;
#pragma unroll
  for (int m = 16; m; m >>= 1) s += __shfl_xor(s, m, 32);
  float mu = s * (1.0f / HH);
  float dx = v.x - mu, dy = v.y - mu, dz = v.z - mu, dw = v.w - mu;
  float q = dx * dx + dy * dy + dz * dz + dw * dw;
#pragma unroll
  for (int m = 16; m; m >>= 1) q += __shfl_xor(q, m, 32);
  float rs = __frsqrt_rn(q * (1.0f / HH) + 1e-5f);
  float4 g = ((const float4*)gamma)[lane];
  float4 be = ((const float4*)beta)[lane];
  float4 o = make_float4(dx * rs * g.x + be.x, dy * rs * g.y + be.y,
                         dz * rs * g.z + be.z, dw * rs * g.w + be.w);
  ((float4*)(out + (size_t)w * HH))[lane] = o;
}

// ---------------------------------------------------------------- launch
extern "C" void kernel_launch(void* const* d_in, const int* in_sizes, int n_in,
                              void* d_out, int out_size, void* d_ws, size_t ws_size,
                              hipStream_t stream) {
  (void)in_sizes; (void)n_in; (void)out_size; (void)ws_size;
  const float* x     = (const float*)d_in[0];
  const int*   src   = (const int*)d_in[1];
  const int*   dst   = (const int*)d_in[2];
  const float* W_in  = (const float*)d_in[3];
  const float* b_in  = (const float*)d_in[4];
  const float* Wb    = (const float*)d_in[5];
  const float* bb    = (const float*)d_in[6];
  const float* gamma = (const float*)d_in[7];
  const float* beta  = (const float*)d_in[8];
  const float* W1    = (const float*)d_in[9];
  const float* b1    = (const float*)d_in[10];
  const float* W2    = (const float*)d_in[11];
  const float* b2    = (const float*)d_in[12];
  float* out = (float*)d_out;

  char* ws = (char*)d_ws;
  size_t off = 0;
  auto alloc = [&](size_t bytes) -> void* {
    void* p = ws + off;
    off = (off + bytes + 255) & ~(size_t)255;
    return p;
  };
  float*  dinv = (float*)alloc((size_t)NN * 4);
  float*  h    = (float*)alloc((size_t)NN * HH * 4);
  float*  tmp  = (float*)alloc((size_t)NN * HH * 4);
  float*  agg  = (float*)alloc((size_t)NN * HH * 4);
  __bf16* abf  = (__bf16*)alloc((size_t)NN * HH * 2);   // bf16 activations
  __bf16* Winb = (__bf16*)alloc((size_t)HH * HH * 2);
  __bf16* Wbb  = (__bf16*)alloc((size_t)KB * HH * HH * 2);
  __bf16* W1b  = (__bf16*)alloc((size_t)HH * HH * 2);
  __bf16* W2b  = (__bf16*)alloc((size_t)HH * 16 * 2);

  const int NH4 = NN * HH / 4;  // float4 count of an activation buffer

  // weights -> bf16 (reuse act2bf; weight sizes are multiples of 4)
  act2bf<<<(HH * HH / 4 + 255) / 256, 256, 0, stream>>>(W_in, Winb, HH * HH / 4);
  act2bf<<<(KB * HH * HH / 4 + 255) / 256, 256, 0, stream>>>(Wb, Wbb, KB * HH * HH / 4);
  act2bf<<<(HH * HH / 4 + 255) / 256, 256, 0, stream>>>(W1, W1b, HH * HH / 4);
  conv_w2pad<<<(HH * 16 + 255) / 256, 256, 0, stream>>>(W2, W2b);

  // degree -> dinv
  deg_init<<<(NN + 255) / 256, 256, 0, stream>>>(dinv, NN);
  deg_edges<<<(EE + 255) / 256, 256, 0, stream>>>(dst, dinv, EE);
  make_dinv<<<(NN + 255) / 256, 256, 0, stream>>>(dinv, NN);

  // input MLP: h = relu(x @ W_in + b_in)
  act2bf<<<NH4 / 256, 256, 0, stream>>>(x, abf, NH4);
  gemm128<<<NN / (16 * MT), 256, 0, stream>>>(abf, Winb, b_in, h, 1);

  // residual GCN layers
  for (int d = 0; d < DEPTH_; ++d) {
    int k = d % KB;
    act2bf<<<NH4 / 256, 256, 0, stream>>>(h, abf, NH4);
    gemm128<<<NN / (16 * MT), 256, 0, stream>>>(abf, Wbb + (size_t)k * HH * HH,
                                                nullptr, tmp, 0);
    agg_init<<<NH4 / 256, 256, 0, stream>>>(tmp, dinv, agg);
    edge_agg<<<EE / 8, 256, 0, stream>>>(tmp, src, dst, dinv, agg, EE);
    residual<<<NH4 / 256, 256, 0, stream>>>(h, agg, bb + (size_t)k * HH);
  }

  // layernorm -> tmp
  layernorm<<<NN / 8, 256, 0, stream>>>(h, gamma, beta, tmp, NN);
  // h1 = relu(tmp @ W1 + b1) -> agg
  act2bf<<<NH4 / 256, 256, 0, stream>>>(tmp, abf, NH4);
  gemm128<<<NN / (16 * MT), 256, 0, stream>>>(abf, W1b, b1, agg, 1);
  // out = agg @ W2 + b2
  act2bf<<<NH4 / 256, 256, 0, stream>>>(agg, abf, NH4);
  gemm_out<<<(NN / 16 + 7) / 8, 256, 0, stream>>>(abf, W2b, b2, out);
}
